// GATModel_33191507263636
// MI455X (gfx1250) — matrix operand order repaired and checked by reading, not compiled, not run on hardware
//
#include <hip/hip_runtime.h>
#include <hip/hip_bf16.h>
#include <math.h>

#define HEADS 4
#define HID 64
#define OUTM 256   // HEADS*HID
#define BN_EPS 1e-5f

typedef __attribute__((ext_vector_type(16))) __bf16 v16bf;
typedef __attribute__((ext_vector_type(16))) unsigned short v16u;
typedef __attribute__((ext_vector_type(8)))  unsigned int   v8u;
typedef __attribute__((ext_vector_type(8)))  float v8f;

// ---------- helpers ----------
__device__ __forceinline__ unsigned short f2bf(float f) {
    unsigned u = __float_as_uint(f);
    u += 0x7FFFu + ((u >> 16) & 1u);   // round-to-nearest-even
    return (unsigned short)(u >> 16);
}

__device__ __forceinline__ void atomicMaxF(float* addr, float val) {
    if (val >= 0.0f) atomicMax((int*)addr, __float_as_int(val));
    else             atomicMin((unsigned int*)addr, __float_as_uint(val));
}

// ---------- generic fill ----------
__global__ void k_fill_f32(float* p, long long n, float v) {
    long long i = (long long)blockIdx.x * blockDim.x + threadIdx.x;
    long long st = (long long)gridDim.x * blockDim.x;
    for (; i < n; i += st) p[i] = v;
}

// ---------- W -> bf16, pre-swizzled into B-fragment layout, zero-padded K ----------
// bswz[(((kbi*16 + ct)*32 + lane)*16 + e] = bf16(W[kbi*32 + (e&7) + (e>>3)*16 + (lane>>4)*8][ct*16 + (lane&15)])
__global__ void k_w_swz(const float* __restrict__ W, unsigned short* __restrict__ bswz, int K) {
    int i = blockIdx.x * blockDim.x + threadIdx.x;
    int KB = (K + 31) >> 5;
    int total = KB << 13;            // KB * 16 * 32 * 16
    if (i >= total) return;
    int e   = i & 15;
    int L   = (i >> 4) & 31;
    int ct  = (i >> 9) & 15;
    int kbi = i >> 13;
    int k = kbi * 32 + (e & 7) + (e >> 3) * 16 + (L >> 4) * 8;
    int n = ct * 16 + (L & 15);
    bswz[i] = (k < K) ? f2bf(W[k * OUTM + n]) : (unsigned short)0;
}

// ---------- A fragment loader: vectorized, compile-time bounds ----------
template<int K>
__device__ __forceinline__ v16bf load_a_frag(const float* __restrict__ xr, int kb, int hi) {
    v16u au;
    float r0[8];
    *(float4*)&r0[0] = *(const float4*)(xr + kb + hi * 8);
    *(float4*)&r0[4] = *(const float4*)(xr + kb + hi * 8 + 4);
#pragma unroll
    for (int e = 0; e < 8; ++e) au[e] = f2bf(r0[e]);
    if (kb + 32 <= K) {              // hi-independent => compile-time under unroll
        float r1[8];
        *(float4*)&r1[0] = *(const float4*)(xr + kb + 16 + hi * 8);
        *(float4*)&r1[4] = *(const float4*)(xr + kb + 16 + hi * 8 + 4);
#pragma unroll
        for (int e = 0; e < 8; ++e) au[8 + e] = f2bf(r1[e]);
    } else {
#pragma unroll
        for (int e = 0; e < 8; ++e) au[8 + e] = 0;
    }
    return __builtin_bit_cast(v16bf, au);
}

// ---------- WMMA GEMM: C[N,256] = X[N,K] * W[K,256]; 2 row-tiles per wave ----------
template<int K>
__global__ __launch_bounds__(256) void k_gemm_wmma(
        const float* __restrict__ X, const unsigned short* __restrict__ bswz,
        float* __restrict__ C, int N) {
    constexpr int KB = (K + 31) / 32;
    int lane = threadIdx.x & 31;
    int wave = (blockIdx.x * blockDim.x + threadIdx.x) >> 5;
    int nwork = (N >> 5) * 16;                 // row-pairs * 16 col tiles
    if (wave >= nwork) return;                 // wave-uniform exit
    int rp = wave >> 4;
    int ct = wave & 15;
    int ml = lane & 15;
    int hi = lane >> 4;
    const float* x0 = X + (size_t)(rp * 32 + ml) * K;
    const float* x1 = x0 + (size_t)16 * K;
    v8f acc0 = {}, acc1 = {};
#pragma unroll
    for (int kbi = 0; kbi < KB; ++kbi) {
        int kb = kbi * 32;
        v8u braw = *(const v8u*)(bswz + (((size_t)(kbi * 16 + ct) * 32 + lane) * 16));
        v16bf b = __builtin_bit_cast(v16bf, braw);
        v16bf a0 = load_a_frag<K>(x0, kb, hi);
        acc0 = __builtin_amdgcn_wmma_f32_16x16x32_bf16(false, a0, false, b,
                                                       (short)0, acc0, false, false);
        v16bf a1 = load_a_frag<K>(x1, kb, hi);
        acc1 = __builtin_amdgcn_wmma_f32_16x16x32_bf16(false, a1, false, b,
                                                       (short)0, acc1, false, false);
    }
    float* c0 = C + (size_t)(rp * 32 + hi * 8) * OUTM + ct * 16 + ml;
#pragma unroll
    for (int i = 0; i < 8; ++i) c0[(size_t)i * OUTM] = acc0[i];
    float* c1 = c0 + (size_t)16 * OUTM;
#pragma unroll
    for (int i = 0; i < 8; ++i) c1[(size_t)i * OUTM] = acc1[i];
}

// ---------- per-node attention logits ----------
__global__ void k_att_node(const float* __restrict__ xh,
                           const float* __restrict__ att_s, const float* __restrict__ att_d,
                           float* __restrict__ a_src, float* __restrict__ a_dst, int N) {
    int i = blockIdx.x * blockDim.x + threadIdx.x;   // n*HEADS + h
    if (i >= N * HEADS) return;
    int n = i >> 2, h = i & 3;
    const float* xr = xh + (long long)n * OUTM + h * HID;
    const float* as = att_s + h * HID;
    const float* ad = att_d + h * HID;
    float s = 0.f, d = 0.f;
    for (int c = 0; c < HID; ++c) { float v = xr[c]; s += v * as[c]; d += v * ad[c]; }
    a_src[i] = s; a_dst[i] = d;
}

// ---------- ce[h] = sum_c W_e[h*64+c]*att_edge[h,c] ----------
__global__ void k_ce(const float* __restrict__ We, const float* __restrict__ att_e,
                     float* __restrict__ ce) {
    int h = threadIdx.x;
    if (h < HEADS) {
        float s = 0.f;
        for (int c = 0; c < HID; ++c) s += We[h * HID + c] * att_e[h * HID + c];
        ce[h] = s;
    }
}

// ---------- self-loop attr: mean of incoming edge attrs ----------
__global__ void k_loop_stats(const int* __restrict__ ei, const float* __restrict__ ea,
                             float* __restrict__ lsum, float* __restrict__ lcnt, int E) {
    int e = blockIdx.x * blockDim.x + threadIdx.x;
    if (e >= E) return;
    int d = ei[E + e];
    atomicAdd(&lsum[d], ea[e]);
    atomicAdd(&lcnt[d], 1.0f);
}
__global__ void k_build_eattr(const float* __restrict__ ea, const float* __restrict__ lsum,
                              const float* __restrict__ lcnt, float* __restrict__ ef,
                              int E, int N) {
    int i = blockIdx.x * blockDim.x + threadIdx.x;
    if (i >= E + N) return;
    ef[i] = (i < E) ? ea[i] : lsum[i - E] / fmaxf(lcnt[i - E], 1.0f);
}

// ---------- edge pass 1: alpha = leaky_relu(...); segment max via atomics ----------
__global__ void k_edge_alpha_max(const int* __restrict__ ei, const float* __restrict__ ef,
                                 const float* __restrict__ a_src, const float* __restrict__ a_dst,
                                 const float* __restrict__ ce,
                                 float* __restrict__ alpha, float* __restrict__ mbuf,
                                 int E, int N) {
    int i = blockIdx.x * blockDim.x + threadIdx.x;   // (e*HEADS + h)
    if (i >= (E + N) * HEADS) return;
    int e = i >> 2, h = i & 3;
    int s, d;
    if (e < E) { s = ei[e]; d = ei[E + e]; } else { s = d = e - E; }
    float a = a_src[s * 4 + h] + a_dst[d * 4 + h] + ef[e] * ce[h];
    a = (a > 0.f) ? a : 0.2f * a;
    alpha[i] = a;
    atomicMaxF(&mbuf[d * 4 + h], a);
}

// ---------- edge pass 2: ex = exp(alpha - max); segment sum ----------
__global__ void k_edge_exp_sum(const int* __restrict__ ei,
                               float* __restrict__ alpha, const float* __restrict__ mbuf,
                               float* __restrict__ den, int E, int N) {
    int i = blockIdx.x * blockDim.x + threadIdx.x;
    if (i >= (E + N) * HEADS) return;
    int e = i >> 2, h = i & 3;
    int d = (e < E) ? ei[E + e] : e - E;
    float m = mbuf[d * 4 + h];
    if (!isfinite(m)) m = 0.f;
    float ex = expf(alpha[i] - m);
    alpha[i] = ex;
    atomicAdd(&den[d * 4 + h], ex);
}

// ---------- edge pass 3: agg[dst] += (ex/den) * xh[src] ----------
__global__ void k_edge_aggregate(const int* __restrict__ ei, const float* __restrict__ alpha,
                                 const float* __restrict__ den, const float* __restrict__ xh,
                                 float* __restrict__ agg, int E, int N) {
    int i = blockIdx.x * blockDim.x + threadIdx.x;   // (e*HEADS + h)
    if (i >= (E + N) * HEADS) return;
    int e = i >> 2, h = i & 3;
    int s, d;
    if (e < E) { s = ei[e]; d = ei[E + e]; } else { s = d = e - E; }
    float w = alpha[i] / fmaxf(den[d * 4 + h], 1e-16f);
    const float4* xv = (const float4*)(xh + (long long)s * OUTM + h * HID);
    float* ap = agg + (long long)d * OUTM + h * HID;
#pragma unroll
    for (int c4 = 0; c4 < HID / 4; ++c4) {
        float4 v = xv[c4];
        atomicAdd(ap + c4 * 4 + 0, w * v.x);
        atomicAdd(ap + c4 * 4 + 1, w * v.y);
        atomicAdd(ap + c4 * 4 + 2, w * v.z);
        atomicAdd(ap + c4 * 4 + 3, w * v.w);
    }
}

// ---------- combine (concat layers): +bias, BN, ELU ----------
__global__ void k_combine_concat(const float* __restrict__ agg, const float* __restrict__ bias,
                                 const float* __restrict__ gamma, const float* __restrict__ beta,
                                 const float* __restrict__ mean, const float* __restrict__ var,
                                 float* __restrict__ out, long long total) {
    long long i = (long long)blockIdx.x * blockDim.x + threadIdx.x;
    if (i >= total) return;
    int j = (int)(i & (OUTM - 1));
    float v = agg[i] + bias[j];
    v = (v - mean[j]) * rsqrtf(var[j] + BN_EPS) * gamma[j] + beta[j];
    out[i] = (v > 0.f) ? v : (expf(v) - 1.f);     // ELU
}

// ---------- combine (mean layer): head mean, +bias, BN ----------
__global__ void k_combine_mean(const float* __restrict__ agg, const float* __restrict__ bias,
                               const float* __restrict__ gamma, const float* __restrict__ beta,
                               const float* __restrict__ mean, const float* __restrict__ var,
                               float* __restrict__ out, int N) {
    int i = blockIdx.x * blockDim.x + threadIdx.x;   // n*HID + c
    if (i >= N * HID) return;
    int n = i >> 6, c = i & 63;
    float v = 0.f;
#pragma unroll
    for (int h = 0; h < HEADS; ++h) v += agg[(long long)n * OUTM + h * HID + c];
    v = v * 0.25f + bias[c];
    v = (v - mean[c]) * rsqrtf(var[c] + BN_EPS) * gamma[c] + beta[c];
    out[i] = v;
}

// ---------- global mean/max pool ----------
__global__ void k_pool_scatter(const float* __restrict__ act, const int* __restrict__ batch,
                               float* __restrict__ psum, float* __restrict__ pmax,
                               float* __restrict__ pcnt, int N) {
    int i = blockIdx.x * blockDim.x + threadIdx.x;   // n*HID + c
    if (i >= N * HID) return;
    int n = i >> 6, c = i & 63;
    int g = batch[n];
    float v = act[i];
    atomicAdd(&psum[g * HID + c], v);
    atomicMaxF(&pmax[g * HID + c], v);
    if (c == 0) atomicAdd(&pcnt[g], 1.0f);
}
__global__ void k_pool_finish(const float* __restrict__ psum, const float* __restrict__ pmax,
                              const float* __restrict__ pcnt, float* __restrict__ hb, int G) {
    int i = blockIdx.x * blockDim.x + threadIdx.x;   // g*HID + c
    if (i >= G * HID) return;
    int g = i >> 6, c = i & 63;
    float mn = psum[i] / fmaxf(pcnt[g], 1.0f);
    float mx = pmax[i];
    if (!isfinite(mx)) mx = 0.f;
    hb[g * 128 + c] = mn;
    hb[g * 128 + 64 + c] = mx;
}

// ---------- tiny dense MLP layer ----------
__global__ void k_mlp(const float* __restrict__ in, const float* __restrict__ W,
                      const float* __restrict__ b, float* __restrict__ out,
                      int G, int K, int J, int relu) {
    int i = blockIdx.x * blockDim.x + threadIdx.x;
    if (i >= G * J) return;
    int g = i / J, j = i % J;
    float acc = b[j];
    for (int k = 0; k < K; ++k) acc += in[g * K + k] * W[k * J + j];
    out[i] = relu ? fmaxf(acc, 0.f) : acc;
}

// =================== host ===================
static inline int cdiv(long long a, long long b) { return (int)((a + b - 1) / b); }

extern "C" void kernel_launch(void* const* d_in, const int* in_sizes, int n_in,
                              void* d_out, int out_size, void* d_ws, size_t ws_size,
                              hipStream_t stream) {
    const float* x     = (const float*)d_in[0];
    const int*   ei    = (const int*)d_in[1];
    const float* ea    = (const float*)d_in[2];
    const int*   batch = (const int*)d_in[3];
    const float* P[36];
    for (int i = 0; i < 36 && (4 + i) < n_in; ++i) P[i] = (const float*)d_in[4 + i];

    const int N = in_sizes[0] / 16;
    const int E = in_sizes[1] / 2;
    const int G = out_size / 2;

    char* w = (char*)d_ws;
    auto carve = [&](size_t bytes) -> void* {
        void* p = (void*)w; w += (bytes + 255) & ~(size_t)255; return p;
    };
    float* bufs[2];
    bufs[0] = (float*)carve((size_t)N * OUTM * 4);
    bufs[1] = (float*)carve((size_t)N * OUTM * 4);
    float* a_src = (float*)carve((size_t)N * HEADS * 4);
    float* a_dst = (float*)carve((size_t)N * HEADS * 4);
    float* mbuf  = (float*)carve((size_t)N * HEADS * 4);
    float* den   = (float*)carve((size_t)N * HEADS * 4);
    float* alpha = (float*)carve((size_t)(E + N) * HEADS * 4);
    float* ef    = (float*)carve((size_t)(E + N) * 4);
    unsigned short* bw = (unsigned short*)carve((size_t)OUTM * OUTM * 2);
    float* lsum  = (float*)carve((size_t)N * 4);
    float* lcnt  = (float*)carve((size_t)N * 4);
    float* ce    = (float*)carve(256);
    float* psum  = (float*)carve((size_t)G * HID * 4);
    float* pmax  = (float*)carve((size_t)G * HID * 4);
    float* pcnt  = (float*)carve((size_t)G * 4);
    float* hbuf  = (float*)carve((size_t)G * 128 * 4);
    float* t1    = (float*)carve((size_t)G * 128 * 4);
    float* t2    = (float*)carve((size_t)G * 64 * 4);

    const int TB = 256;
    const int FG = 2048;

    // ---- self-loop edge attrs (layer-invariant) ----
    k_fill_f32<<<FG, TB, 0, stream>>>(lsum, N, 0.f);
    k_fill_f32<<<FG, TB, 0, stream>>>(lcnt, N, 0.f);
    k_loop_stats<<<cdiv(E, TB), TB, 0, stream>>>(ei, ea, lsum, lcnt, E);
    k_build_eattr<<<cdiv(E + N, TB), TB, 0, stream>>>(ea, lsum, lcnt, ef, E, N);

    const float* cur = x;
    for (int L = 0; L < 3; ++L) {
        const float** pp = P + L * 10;
        float* xh  = bufs[L & 1];
        float* agg = bufs[1 - (L & 1)];    // dead layer-input buffer (stream-ordered reuse)

        // GEMM: xh = cur @ W  (bf16 WMMA, pre-swizzled B)
        int K = (L == 0) ? 16 : OUTM;
        int KBp = (K + 31) >> 5;
        k_w_swz<<<cdiv((long long)KBp << 13, TB), TB, 0, stream>>>(pp[0], bw, K);
        int nwork = (N / 32) * 16;         // waves
        if (L == 0)
            k_gemm_wmma<16><<<cdiv(nwork, 8), TB, 0, stream>>>(cur, bw, xh, N);
        else
            k_gemm_wmma<OUTM><<<cdiv(nwork, 8), TB, 0, stream>>>(cur, bw, xh, N);

        // attention logits
        k_att_node<<<cdiv((long long)N * HEADS, TB), TB, 0, stream>>>(xh, pp[2], pp[3], a_src, a_dst, N);
        k_ce<<<1, 32, 0, stream>>>(pp[1], pp[4], ce);

        // init per-node reductions + aggregate target
        k_fill_f32<<<FG, TB, 0, stream>>>(mbuf, (long long)N * HEADS, -INFINITY);
        k_fill_f32<<<FG, TB, 0, stream>>>(den,  (long long)N * HEADS, 0.f);
        k_fill_f32<<<FG, TB, 0, stream>>>(agg,  (long long)N * OUTM, 0.f);

        long long eh = (long long)(E + N) * HEADS;
        k_edge_alpha_max<<<cdiv(eh, TB), TB, 0, stream>>>(ei, ef, a_src, a_dst, ce, alpha, mbuf, E, N);
        k_edge_exp_sum<<<cdiv(eh, TB), TB, 0, stream>>>(ei, alpha, mbuf, den, E, N);
        k_edge_aggregate<<<cdiv(eh, TB), TB, 0, stream>>>(ei, alpha, den, xh, agg, E, N);

        if (L < 2) {
            k_combine_concat<<<cdiv((long long)N * OUTM, TB), TB, 0, stream>>>(
                agg, pp[5], pp[6], pp[7], pp[8], pp[9], xh, (long long)N * OUTM);
        } else {
            k_combine_mean<<<cdiv((long long)N * HID, TB), TB, 0, stream>>>(
                agg, pp[5], pp[6], pp[7], pp[8], pp[9], xh, N);
        }
        cur = xh;
    }
    const float* act = bufs[0];   // layer-2 output, [N, 64]

    // ---- pooling ----
    k_fill_f32<<<FG, TB, 0, stream>>>(psum, (long long)G * HID, 0.f);
    k_fill_f32<<<FG, TB, 0, stream>>>(pmax, (long long)G * HID, -INFINITY);
    k_fill_f32<<<FG, TB, 0, stream>>>(pcnt, G, 0.f);
    k_pool_scatter<<<cdiv((long long)N * HID, TB), TB, 0, stream>>>(act, batch, psum, pmax, pcnt, N);
    k_pool_finish<<<cdiv((long long)G * HID, TB), TB, 0, stream>>>(psum, pmax, pcnt, hbuf, G);

    // ---- MLP head ----
    const float** M = P + 30;
    k_mlp<<<cdiv((long long)G * 128, TB), TB, 0, stream>>>(hbuf, M[0], M[1], t1, G, 128, 128, 1);
    k_mlp<<<cdiv((long long)G * 64, TB), TB, 0, stream>>>(t1, M[2], M[3], t2, G, 128, 64, 1);
    k_mlp<<<cdiv((long long)G * 2, TB), TB, 0, stream>>>(t2, M[4], M[5], (float*)d_out, G, 64, 2, 0);
}
